// Transformer_5574867550579
// MI455X (gfx1250) — matrix operand order repaired and checked by reading, not compiled, not run on hardware
//
#include <hip/hip_runtime.h>
#include <hip/hip_bf16.h>
#include <math.h>

// ---- model constants (match reference) ----
#define B_   4
#define LS_  512
#define LT_  512
#define V_   32000
#define H_   1024
#define G_   4
#define NH_  16
#define HD_  64
#define KVD_ 256
#define FF_  4096
#define LE_  4
#define LD_  4
#define EPS_ 1.1920929e-07f

typedef __attribute__((ext_vector_type(16))) __bf16 v16bf;
typedef __attribute__((ext_vector_type(8)))  float  v8f;

__device__ __forceinline__ unsigned int pack2_bf16(float a, float b) {
  unsigned int ua = __float_as_uint(a);
  unsigned int ub = __float_as_uint(b);
  ua = (ua + 0x7FFFu + ((ua >> 16) & 1u)) >> 16;   // RNE
  ub = (ub + 0x7FFFu + ((ub >> 16) & 1u)) >> 16;
  return ua | (ub << 16);
}
__device__ __forceinline__ unsigned short f2bf(float a) {
  unsigned int u = __float_as_uint(a);
  u = (u + 0x7FFFu + ((u >> 16) & 1u)) >> 16;
  return (unsigned short)u;
}

// CDNA5 async global->LDS copy (ASYNCcnt-tracked, no VGPR data path)
__device__ __forceinline__ void async_cp_b128(unsigned lds_addr, const void* gptr) {
  asm volatile("global_load_async_to_lds_b128 %0, %1, off"
               :: "v"(lds_addr), "v"((unsigned long long)(size_t)gptr)
               : "memory");
}
__device__ __forceinline__ void wait_async0() {
  asm volatile("s_wait_asynccnt 0x0" ::: "memory");
}

// ---- WMMA GEMM: out[M,N] = A[M,K](bf16) @ W[N,K](bf16)^T (+bias)(+GELU)(+residual) ----
#define BM 128
#define BN 128
#define BK 32
#define PADB 80                 // LDS row stride in bytes (64B data + 16B pad), 16B aligned
#define TILEB (BM * PADB)       // 10240 B per tile

union Frag { v16bf v; uint4 q[2]; };

__global__ __launch_bounds__(256) void gemm_bf16_wmma(
    const unsigned short* __restrict__ A, const unsigned short* __restrict__ W,
    const float* __restrict__ bias, const float* __restrict__ residual,
    void* __restrict__ outp, int M, int N, int K, int do_gelu, int store_bf16)
{
  __shared__ __align__(16) char smem[2 * 2 * TILEB];   // [buf][A|B] double-buffered
  const int tid  = threadIdx.x;
  const int bm0  = blockIdx.y * BM;
  const int bn0  = blockIdx.x * BN;
  const int wave = tid >> 5, lane = tid & 31;
  const int wr = wave >> 1, wc = wave & 1;             // 4x2 wave grid -> 32x64 per wave
  const int mlane = lane & 15, khalf = lane >> 4;

  const unsigned lds0 = (unsigned)(size_t)(&smem[0]);

  v8f acc[2][4];
#pragma unroll
  for (int i = 0; i < 2; ++i)
#pragma unroll
    for (int j = 0; j < 4; ++j)
#pragma unroll
      for (int e = 0; e < 8; ++e) acc[i][j][e] = 0.0f;

  const int nk = K / BK;
  // stage tile kt into buffer buf: 1024 b128 chunks, 4 per thread (2 A, 2 B)
  auto stage = [&](int buf, int k0) {
#pragma unroll
    for (int s = 0; s < 2; ++s) {
      int c   = tid + 256 * s;         // 0..511
      int row = c >> 2;                // 4 chunks of 16B per 64B row
      int cb  = (c & 3) << 4;
      unsigned la = lds0 + (unsigned)(buf * 2 * TILEB + row * PADB + cb);
      async_cp_b128(la, (const char*)A + (((size_t)(bm0 + row) * K + k0) << 1) + cb);
      async_cp_b128(la + TILEB, (const char*)W + (((size_t)(bn0 + row) * K + k0) << 1) + cb);
    }
  };

  stage(0, 0);
  for (int kt = 0; kt < nk; ++kt) {
    wait_async0();          // my tile-kt copies complete
    __syncthreads();        // everyone's complete; prev compute done -> other buffer reusable
    if (kt + 1 < nk) stage((kt + 1) & 1, (kt + 1) * BK);

    const char* bufA = smem + (kt & 1) * 2 * TILEB;
    const char* bufB = bufA + TILEB;
    Frag af[2], bfr[4];
#pragma unroll
    for (int i = 0; i < 2; ++i) {      // A 16x32 frag: two 16B chunks per lane
      int row = wr * 32 + i * 16 + mlane;
      const char* p = bufA + row * PADB + (khalf << 4);
      af[i].q[0] = *(const uint4*)(p);
      af[i].q[1] = *(const uint4*)(p + 32);
    }
#pragma unroll
    for (int j = 0; j < 4; ++j) {      // B 32x16 frag: two 16B chunks per lane
      int row = wc * 64 + j * 16 + mlane;
      const char* p = bufB + row * PADB + (khalf << 4);
      bfr[j].q[0] = *(const uint4*)(p);
      bfr[j].q[1] = *(const uint4*)(p + 32);
    }
#pragma unroll
    for (int i = 0; i < 2; ++i)
#pragma unroll
      for (int j = 0; j < 4; ++j)
        acc[i][j] = __builtin_amdgcn_wmma_f32_16x16x32_bf16(
            false, af[i].v, false, bfr[j].v, (short)0, acc[i][j], false, false);
  }

  // epilogue (C/D layout: n = lane&15, m = r + 8*(lane>>4))
#pragma unroll
  for (int j = 0; j < 4; ++j) {
    int col = bn0 + wc * 64 + j * 16 + mlane;
    float bval = bias ? bias[col] : 0.0f;
#pragma unroll
    for (int i = 0; i < 2; ++i) {
#pragma unroll
      for (int r = 0; r < 8; ++r) {
        int row = bm0 + wr * 32 + i * 16 + r + (khalf << 3);
        float v = acc[i][j][r] + bval;
        if (do_gelu) v = 0.5f * v * (1.0f + erff(v * 0.70710678118654752f));
        if (residual) v += residual[(size_t)row * N + col];
        if (store_bf16) ((unsigned short*)outp)[(size_t)row * N + col] = f2bf(v);
        else            ((float*)outp)[(size_t)row * N + col] = v;
      }
    }
  }
}

// ---- f32 -> bf16 bulk convert (weights / residual streams) ----
__global__ __launch_bounds__(256) void conv_bf16_k(const float* __restrict__ in,
    unsigned short* __restrict__ out, long long n2)   // n2 = n/2 pairs
{
  long long i = (long long)blockIdx.x * 256 + threadIdx.x;
  const long long stride = (long long)gridDim.x * 256;
  for (; i < n2; i += stride) {
    float2 v = ((const float2*)in)[i];
    ((unsigned int*)out)[i] = pack2_bf16(v.x, v.y);
  }
}

// ---- reductions (wave32) ----
__device__ __forceinline__ float waveSum(float v) {
#pragma unroll
  for (int o = 16; o > 0; o >>= 1) v += __shfl_xor(v, o, 32);
  return v;
}
__device__ __forceinline__ float waveMax(float v) {
#pragma unroll
  for (int o = 16; o > 0; o >>= 1) v = fmaxf(v, __shfl_xor(v, o, 32));
  return v;
}

// ---- RMSNorm: out(bf16) = x * rsqrt(mean(x^2)+eps) * w ----
__global__ __launch_bounds__(256) void rmsnorm_k(const float* __restrict__ x,
    const float* __restrict__ w, unsigned short* __restrict__ out)
{
  __shared__ float red[8];
  const size_t row = blockIdx.x;
  const float* xr = x + row * H_;
  float ss = 0.f;
  for (int c = threadIdx.x; c < H_; c += 256) { float v = xr[c]; ss += v * v; }
  ss = waveSum(ss);
  if ((threadIdx.x & 31) == 0) red[threadIdx.x >> 5] = ss;
  __syncthreads();
  float tot = 0.f;
#pragma unroll
  for (int i = 0; i < 8; ++i) tot += red[i];
  const float nrm = rsqrtf(tot * (1.0f / H_) + EPS_);
  unsigned short* orow = out + row * H_;
  for (int c = threadIdx.x; c < H_; c += 256) orow[c] = f2bf(xr[c] * nrm * w[c]);
}

// ---- embedding + sinusoidal positional encoding (f32 residual stream) ----
__global__ __launch_bounds__(256) void embed_pe_k(const int* __restrict__ toks,
    const float* __restrict__ emb, float* __restrict__ out, int L)
{
  const int pos = blockIdx.x, b = blockIdx.y;
  const int tok = toks[b * L + pos];
  const float c0 = -9.210340371976184f / (float)H_;  // -ln(10000)/H
  float* orow = out + ((size_t)b * L + pos) * H_;
  const float* erow = emb + (size_t)tok * H_;
  for (int c = threadIdx.x; c < H_; c += 256) {
    float ang = (float)pos * expf(c0 * (float)(c & ~1));
    float pe = (c & 1) ? cosf(ang) : sinf(ang);
    orow[c] = erow[c] + pe;
  }
}

// ---- attention: one block per (b,h,q); analytic masks; bf16 output ----
__global__ __launch_bounds__(256) void attn_k(const float* __restrict__ Q,
    const float* __restrict__ Km, const float* __restrict__ Vm,
    unsigned short* __restrict__ O, int lq, int lk, int causal, int pad_start)
{
  __shared__ float qrow[HD_];
  __shared__ float sc[512];
  __shared__ float red[8];
  __shared__ float obuf[4][HD_];
  const int tid = threadIdx.x;
  const int qi = blockIdx.x, h = blockIdx.y, b = blockIdx.z;
  const int g = h / (NH_ / G_);
  const float* qp = Q + ((size_t)b * lq + qi) * H_ + h * HD_;
  if (tid < HD_) qrow[tid] = qp[tid];
  __syncthreads();
  for (int j = tid; j < lk; j += 256) {
    float s;
    if ((causal && j > qi) || j >= pad_start) s = -INFINITY;
    else {
      const float* kp = Km + ((size_t)b * lk + j) * KVD_ + g * HD_;
      float d = 0.f;
#pragma unroll
      for (int c = 0; c < HD_; ++c) d += qrow[c] * kp[c];
      s = d * 0.125f;
    }
    sc[j] = s;
  }
  __syncthreads();
  float m = -INFINITY;
  for (int j = tid; j < lk; j += 256) m = fmaxf(m, sc[j]);
  m = waveMax(m);
  if ((tid & 31) == 0) red[tid >> 5] = m;
  __syncthreads();
  float mm = -INFINITY;
#pragma unroll
  for (int i = 0; i < 8; ++i) mm = fmaxf(mm, red[i]);
  __syncthreads();
  float sum = 0.f;
  for (int j = tid; j < lk; j += 256) { float p = expf(sc[j] - mm); sc[j] = p; sum += p; }
  sum = waveSum(sum);
  __syncthreads();
  if ((tid & 31) == 0) red[tid >> 5] = sum;
  __syncthreads();
  float tot = 0.f;
#pragma unroll
  for (int i = 0; i < 8; ++i) tot += red[i];
  const float inv = 1.0f / tot;
  const int d = tid & (HD_ - 1), part = tid >> 6;
  float a = 0.f;
  for (int j = part; j < lk; j += 4)
    a += sc[j] * Vm[((size_t)b * lk + j) * KVD_ + g * HD_ + d];
  obuf[part][d] = a;
  __syncthreads();
  if (tid < HD_) {
    float o = (obuf[0][tid] + obuf[1][tid] + obuf[2][tid] + obuf[3][tid]) * inv;
    O[((size_t)b * lq + qi) * H_ + h * HD_ + tid] = f2bf(o);
  }
}

// ---- orchestration ----
extern "C" void kernel_launch(void* const* d_in, const int* in_sizes, int n_in,
                              void* d_out, int out_size, void* d_ws, size_t ws_size,
                              hipStream_t stream) {
  (void)in_sizes; (void)n_in; (void)out_size; (void)ws_size;
  const int*   src = (const int*)d_in[0];
  const int*   tgt = (const int*)d_in[1];
  const float* emb = (const float*)d_in[4];
  const float* lm  = (const float*)d_in[5];
  const float* e_qw = (const float*)d_in[6];
  const float* e_qb = (const float*)d_in[7];
  const float* e_kw = (const float*)d_in[8];
  const float* e_kb = (const float*)d_in[9];
  const float* e_vw = (const float*)d_in[10];
  const float* e_vb = (const float*)d_in[11];
  const float* e_ow = (const float*)d_in[12];
  const float* e_ob = (const float*)d_in[13];
  const float* e_rn1 = (const float*)d_in[14];
  const float* e_w1 = (const float*)d_in[15];
  const float* e_b1 = (const float*)d_in[16];
  const float* e_w2 = (const float*)d_in[17];
  const float* e_b2 = (const float*)d_in[18];
  const float* e_rn2 = (const float*)d_in[19];
  const float* s_qw = (const float*)d_in[20];
  const float* s_qb = (const float*)d_in[21];
  const float* s_kw = (const float*)d_in[22];
  const float* s_kb = (const float*)d_in[23];
  const float* s_vw = (const float*)d_in[24];
  const float* s_vb = (const float*)d_in[25];
  const float* s_ow = (const float*)d_in[26];
  const float* s_ob = (const float*)d_in[27];
  const float* d_rn1 = (const float*)d_in[28];
  const float* c_qw = (const float*)d_in[29];
  const float* c_qb = (const float*)d_in[30];
  const float* c_kw = (const float*)d_in[31];
  const float* c_kb = (const float*)d_in[32];
  const float* c_vw = (const float*)d_in[33];
  const float* c_vb = (const float*)d_in[34];
  const float* c_ow = (const float*)d_in[35];
  const float* c_ob = (const float*)d_in[36];
  const float* d_rn2 = (const float*)d_in[37];
  const float* d_w1 = (const float*)d_in[38];
  const float* d_b1 = (const float*)d_in[39];
  const float* d_w2 = (const float*)d_in[40];
  const float* d_b2 = (const float*)d_in[41];
  const float* d_rn3 = (const float*)d_in[42];

  // workspace carve-up (bytes)
  const size_t TOK = (size_t)B_ * LS_ * H_;      // 2,097,152 elements
  char* w8 = (char*)d_ws;
  float* bx = (float*)(w8);                       // enc stream / enc_out   8MB
  float* by = (float*)(w8 + (8u << 20));          // dec stream             8MB
  float* bq = (float*)(w8 + (16u << 20));         // Q (f32 for attn)       8MB
  float* bk = (float*)(w8 + (24u << 20));         // K                      2MB
  float* bv = (float*)(w8 + (26u << 20));         // V                      2MB
  unsigned short* bn  = (unsigned short*)(w8 + (28u << 20)); // rmsnorm bf16 4MB
  unsigned short* ba  = (unsigned short*)(w8 + (32u << 20)); // attn out bf16 4MB
  unsigned short* bh  = (unsigned short*)(w8 + (36u << 20)); // FFN hidden bf16 16MB
  unsigned short* bxb = (unsigned short*)(w8 + (52u << 20)); // enc_out bf16 4MB
  unsigned short* byb = (unsigned short*)(w8 + (56u << 20)); // dec stream bf16 4MB
  unsigned short* wb  = (unsigned short*)(w8 + (60u << 20)); // weight bf16 cache 64MB

  const int M = B_ * LS_;                          // 2048
  dim3 blk(256);
  dim3 gH(H_ / BN, M / BM), gKV(KVD_ / BN, M / BM), gFF(FF_ / BN, M / BM), gV(V_ / BN, M / BM);
  const long long nQ = (long long)H_ * H_ / 2, nKV = (long long)KVD_ * H_ / 2,
                  nF = (long long)FF_ * H_ / 2, nV = (long long)V_ * H_ / 2, nT = (long long)TOK / 2;
#define CONV(srcp, n) conv_bf16_k<<<2048, blk, 0, stream>>>((srcp), wb, (n))

  // ===== encoder =====
  embed_pe_k<<<dim3(LS_, B_), blk, 0, stream>>>(src, emb, bx, LS_);
  for (int i = 0; i < LE_; ++i) {
    rmsnorm_k<<<M, blk, 0, stream>>>(bx, e_rn1 + (size_t)i * H_, bn);
    CONV(e_qw + (size_t)i * H_ * H_, nQ);
    gemm_bf16_wmma<<<gH, blk, 0, stream>>>(bn, wb, e_qb + (size_t)i * H_, nullptr, bq, M, H_, H_, 0, 0);
    CONV(e_kw + (size_t)i * KVD_ * H_, nKV);
    gemm_bf16_wmma<<<gKV, blk, 0, stream>>>(bn, wb, e_kb + (size_t)i * KVD_, nullptr, bk, M, KVD_, H_, 0, 0);
    CONV(e_vw + (size_t)i * KVD_ * H_, nKV);
    gemm_bf16_wmma<<<gKV, blk, 0, stream>>>(bn, wb, e_vb + (size_t)i * KVD_, nullptr, bv, M, KVD_, H_, 0, 0);
    attn_k<<<dim3(LS_, NH_, B_), blk, 0, stream>>>(bq, bk, bv, ba, LS_, LS_, 0, LS_);
    CONV(e_ow + (size_t)i * H_ * H_, nQ);
    gemm_bf16_wmma<<<gH, blk, 0, stream>>>(ba, wb, e_ob + (size_t)i * H_, bx, bx, M, H_, H_, 0, 0);
    rmsnorm_k<<<M, blk, 0, stream>>>(bx, e_rn2 + (size_t)i * H_, bn);
    CONV(e_w1 + (size_t)i * FF_ * H_, nF);
    gemm_bf16_wmma<<<gFF, blk, 0, stream>>>(bn, wb, e_b1 + (size_t)i * FF_, nullptr, bh, M, FF_, H_, 1, 1);
    CONV(e_w2 + (size_t)i * H_ * FF_, nF);
    gemm_bf16_wmma<<<gH, blk, 0, stream>>>(bh, wb, e_b2 + (size_t)i * H_, bx, bx, M, H_, FF_, 0, 0);
  }
  conv_bf16_k<<<2048, blk, 0, stream>>>(bx, bxb, nT);   // enc_out bf16 for cross K/V
  // ===== decoder =====
  embed_pe_k<<<dim3(LT_, B_), blk, 0, stream>>>(tgt, emb, by, LT_);
  for (int i = 0; i < LD_; ++i) {
    // self-attention (causal + padding at LT-64)
    rmsnorm_k<<<M, blk, 0, stream>>>(by, d_rn1 + (size_t)i * H_, bn);
    CONV(s_qw + (size_t)i * H_ * H_, nQ);
    gemm_bf16_wmma<<<gH, blk, 0, stream>>>(bn, wb, s_qb + (size_t)i * H_, nullptr, bq, M, H_, H_, 0, 0);
    CONV(s_kw + (size_t)i * KVD_ * H_, nKV);
    gemm_bf16_wmma<<<gKV, blk, 0, stream>>>(bn, wb, s_kb + (size_t)i * KVD_, nullptr, bk, M, KVD_, H_, 0, 0);
    CONV(s_vw + (size_t)i * KVD_ * H_, nKV);
    gemm_bf16_wmma<<<gKV, blk, 0, stream>>>(bn, wb, s_vb + (size_t)i * KVD_, nullptr, bv, M, KVD_, H_, 0, 0);
    attn_k<<<dim3(LT_, NH_, B_), blk, 0, stream>>>(bq, bk, bv, ba, LT_, LT_, 1, LT_ - 64);
    CONV(s_ow + (size_t)i * H_ * H_, nQ);
    gemm_bf16_wmma<<<gH, blk, 0, stream>>>(ba, wb, s_ob + (size_t)i * H_, by, by, M, H_, H_, 0, 0);
    // cross-attention (K/V from raw encoder output)
    rmsnorm_k<<<M, blk, 0, stream>>>(by, d_rn2 + (size_t)i * H_, bn);
    CONV(c_qw + (size_t)i * H_ * H_, nQ);
    gemm_bf16_wmma<<<gH, blk, 0, stream>>>(bn, wb, c_qb + (size_t)i * H_, nullptr, bq, M, H_, H_, 0, 0);
    CONV(c_kw + (size_t)i * KVD_ * H_, nKV);
    gemm_bf16_wmma<<<gKV, blk, 0, stream>>>(bxb, wb, c_kb + (size_t)i * KVD_, nullptr, bk, M, KVD_, H_, 0, 0);
    CONV(c_vw + (size_t)i * KVD_ * H_, nKV);
    gemm_bf16_wmma<<<gKV, blk, 0, stream>>>(bxb, wb, c_vb + (size_t)i * KVD_, nullptr, bv, M, KVD_, H_, 0, 0);
    attn_k<<<dim3(LT_, NH_, B_), blk, 0, stream>>>(bq, bk, bv, ba, LT_, LS_, 0, LS_);
    CONV(c_ow + (size_t)i * H_ * H_, nQ);
    gemm_bf16_wmma<<<gH, blk, 0, stream>>>(ba, wb, c_ob + (size_t)i * H_, by, by, M, H_, H_, 0, 0);
    // FFN
    rmsnorm_k<<<M, blk, 0, stream>>>(by, d_rn3 + (size_t)i * H_, bn);
    CONV(d_w1 + (size_t)i * FF_ * H_, nF);
    gemm_bf16_wmma<<<gFF, blk, 0, stream>>>(bn, wb, d_b1 + (size_t)i * FF_, nullptr, bh, M, FF_, H_, 1, 1);
    CONV(d_w2 + (size_t)i * H_ * FF_, nF);
    gemm_bf16_wmma<<<gH, blk, 0, stream>>>(bh, wb, d_b2 + (size_t)i * H_, by, by, M, H_, FF_, 0, 0);
  }
  // ===== LM head =====
  conv_bf16_k<<<2048, blk, 0, stream>>>(by, byb, nT);
  conv_bf16_k<<<4096, blk, 0, stream>>>(lm, wb, nV);
  gemm_bf16_wmma<<<gV, blk, 0, stream>>>(byb, wb, nullptr, nullptr, (float*)d_out, M, V_, H_, 0, 0);
#undef CONV
}
